// SiAttention_33139967655992
// MI455X (gfx1250) — matrix operand order repaired and checked by reading, not compiled, not run on hardware
//
#include <hip/hip_runtime.h>

typedef _Float16 half_t;
typedef __attribute__((ext_vector_type(16))) _Float16 v16h;
typedef __attribute__((ext_vector_type(8)))  _Float16 h8v;
typedef __attribute__((ext_vector_type(8)))  float    v8f;

#define B_   8
#define N_   3136
#define C_   512
#define NH_  8
#define HD_  64
#define H_   56
#define W_   56
#define KS_  5
#define FF_  4096
#define MTOT (B_*N_)   /* 25088 */

#define USE_ASYNC_LDS 1

#if defined(__has_builtin)
#if __has_builtin(__builtin_amdgcn_tensor_load_to_lds) && __has_builtin(__builtin_amdgcn_s_wait_tensorcnt)
#define HAVE_TDM 1
#endif
#endif
#ifndef HAVE_TDM
#define HAVE_TDM 0
#endif

// ---------------- workspace layout (bytes) ----------------
static constexpr size_t OFF_XH   = 0;
static constexpr size_t SZ_XH    = (size_t)MTOT*C_*2;
static constexpr size_t OFF_QGKV = OFF_XH + SZ_XH;                  // qgkv f32 (later hh f16, same size)
static constexpr size_t SZ_QGKV  = (size_t)MTOT*4*C_*4;
static constexpr size_t OFF_WCAT = OFF_QGKV + SZ_QGKV;              // [Wqg|Wkv]^T f16 (2048 x 512)
static constexpr size_t SZ_WCAT  = (size_t)C_*4*C_*2;
static constexpr size_t OFF_W1T  = OFF_WCAT + SZ_WCAT;              // W1^T f16 (4096 x 512)
static constexpr size_t SZ_W1T   = (size_t)C_*FF_*2;
static constexpr size_t OFF_W2T  = OFF_W1T + SZ_W1T;                // W2^T f16 (512 x 4096)
static constexpr size_t SZ_W2T   = (size_t)FF_*C_*2;
static constexpr size_t OFF_QF   = OFF_W2T + SZ_W2T;                // qf f16 (B,NH,N,128)
static constexpr size_t SZ_QF    = (size_t)B_*NH_*N_*128*2;
static constexpr size_t OFF_KCT  = OFF_QF + SZ_QF;                  // kc^T f16 (B,NH,128,N)
static constexpr size_t OFF_VHT  = OFF_KCT + SZ_QF;                 // v^T f16 (B,NH,64,N)
static constexpr size_t SZ_VHT   = SZ_XH;
static constexpr size_t OFF_KM   = OFF_VHT + SZ_VHT;                // km f32
static constexpr size_t SZ_KM    = (size_t)B_*NH_*128*4;
static constexpr size_t OFF_KVT  = OFF_KM + SZ_KM;                  // kvagg^T f16 (B,NH,64,128)
static constexpr size_t SZ_KVT   = (size_t)B_*NH_*64*128*2;
static constexpr size_t OFF_Z    = OFF_KVT + SZ_KVT;                // z f32 (B,NH,N,2)
static constexpr size_t SZ_Z     = (size_t)B_*NH_*N_*2*4;
static constexpr size_t OFF_ATTN = OFF_Z + SZ_Z;                    // attn f32 -> x2 f32 in place
static constexpr size_t SZ_ATTN  = (size_t)MTOT*C_*4;
static constexpr size_t OFF_VOUT = OFF_ATTN + SZ_ATTN;              // vout f32 -> ff f32 later

// ---------------- async global->LDS lane copy (ASYNCcnt path) ----------------
__device__ __forceinline__ void cp_lds16(void* lds_dst, const void* gsrc) {
#if USE_ASYNC_LDS
  uint32_t l = (uint32_t)(uintptr_t)lds_dst;
  uint64_t g = (uint64_t)(uintptr_t)gsrc;
  asm volatile("global_load_async_to_lds_b128 %0, %1, off" :: "v"(l), "v"(g) : "memory");
#else
  *(h8v*)lds_dst = *(const h8v*)gsrc;
#endif
}
#if USE_ASYNC_LDS
#define CP_WAIT(n) asm volatile("s_wait_asynccnt " #n ::: "memory")
#else
#define CP_WAIT(n) do {} while (0)
#endif

// ---------------- Tensor Data Mover descriptor load (TENSORcnt path) ----------------
#if HAVE_TDM
typedef __attribute__((ext_vector_type(4))) unsigned int u32x4;
typedef __attribute__((ext_vector_type(8))) int i32x8;
typedef __attribute__((ext_vector_type(4))) int i32x4;

// 2D tile (tile_d0 elems/row x tile_d1 rows) of 2-byte elems, global row stride
// stride_elems, with LDS padding: after 2^(pad_int+1) DWORDs insert (pad_amt+1) DWORDs.
__device__ __forceinline__ void tdm_load_2d(void* lds_dst, const void* gsrc,
                                            unsigned tile_d0, unsigned tile_d1,
                                            unsigned stride_elems,
                                            unsigned pad_int, unsigned pad_amt) {
  uint32_t laddr = (uint32_t)(uintptr_t)lds_dst;
  uint64_t gaddr = (uint64_t)(uintptr_t)gsrc;
  u32x4 g0;
  g0[0] = 1u;                                      // count=1 (valid), no gather
  g0[1] = laddr;                                   // lds_addr
  g0[2] = (uint32_t)gaddr;                         // global_addr[31:0]
  g0[3] = (uint32_t)(gaddr >> 32) | (2u << 30);    // global_addr[56:32] | type=2 ("image")
  i32x8 g1;
  g1[0] = (int)((1u << 16) | (1u << 20) | (pad_int << 22) | (pad_amt << 25)); // 2B elems, pad_enable
  g1[1] = (int)((tile_d0 & 0xFFFFu) << 16);                       // tensor_dim0 lo16 (== tile_d0)
  g1[2] = (int)((tile_d0 >> 16) | ((tile_d1 & 0xFFFFu) << 16));   // tensor_dim0 hi | tensor_dim1 lo
  g1[3] = (int)((tile_d1 >> 16) | ((tile_d0 & 0xFFFFu) << 16));   // tensor_dim1 hi | tile_dim0
  g1[4] = (int)(tile_d1 & 0xFFFFu);                               // tile_dim1 (tile_dim2 = 0)
  g1[5] = (int)stride_elems;                                      // tensor_dim0_stride[31:0]
  g1[6] = 0;
  g1[7] = 0;
  i32x4 z4 = {0, 0, 0, 0};
  i32x8 z8 = {0, 0, 0, 0, 0, 0, 0, 0};
  __builtin_amdgcn_tensor_load_to_lds(g0, g1, z4, z4, z8, 0);   // 6-arg form (clang-23)
}
#endif

// ---------------- WMMA helpers (gfx1250, wave32) ----------------
__device__ __forceinline__ v8f wmma16(const v16h& a, const v16h& b, const v8f& c) {
  return __builtin_amdgcn_wmma_f32_16x16x32_f16(false, a, false, b, (short)0, c, false, false);
}

// A fragment from row-major LDS tile (no permutation): lane m=lane&15, halves at
// k0+kh*8 and k0+16+kh*8 (each 16B aligned -> ds_load_b128 pair)
__device__ __forceinline__ v16h frag_a16(const half_t* lds, int ld, int row0, int k0, int lane) {
  int m = lane & 15, kh = (lane >> 4) & 1;
  const half_t* p = lds + (size_t)(row0 + m) * ld + k0 + kh * 8;
  h8v lo = *(const h8v*)p;
  h8v hi = *(const h8v*)(p + 16);
  v16h r;
#pragma unroll
  for (int j = 0; j < 8; ++j) { r[j] = lo[j]; r[8 + j] = hi[j]; }
  return r;
}
// B fragment from column-major (B^T) LDS tile: lane holds column col0+n, halves at
// k0+kg*8 and k0+16+kg*8 (same k order as frag_a16)
__device__ __forceinline__ v16h frag_bT(const half_t* lds, int ld, int k0, int col0, int lane) {
  int n = lane & 15, kg = (lane >> 4) & 1;
  const half_t* p = lds + (size_t)(col0 + n) * ld + k0 + kg * 8;
  h8v lo = *(const h8v*)p;
  h8v hi = *(const h8v*)(p + 16);
  v16h r;
#pragma unroll
  for (int j = 0; j < 8; ++j) { r[j] = lo[j]; r[8 + j] = hi[j]; }
  return r;
}

// ---------------- generic GEMM, TDM-staged, double-buffered ----------------
// Block tile 128x128, wave tile 32x64. EPI: 0 f32, 1 bias+relu f16, 2 bias f32
template <int EPI>
__global__ __launch_bounds__(256) void gemm_f16_k(const half_t* __restrict__ A,
                                                  const half_t* __restrict__ BT,
                                                  void* __restrict__ Out,
                                                  const float* __restrict__ bias,
                                                  int M, int Nn, int K) {
  constexpr int LDA = 48, LDB = 48;      // 32 data + 16 pad halfs (TDM: pad 8 DW per 16 DW)
  constexpr int ASZ = 128 * LDA, BSZ = 128 * LDB;
  __shared__ half_t As[2 * ASZ];
  __shared__ half_t Bs[2 * BSZ];
  const int tid  = threadIdx.x;
  const int lane = tid & 31, wid = tid >> 5;
  const int bm = blockIdx.x * 128, bn = blockIdx.y * 128;
  const int wm = (wid >> 1) * 32, wn = (wid & 1) * 64;

  auto stage = [&](int buf, int kb) {
#if HAVE_TDM
    if (wid == 0) {
      tdm_load_2d(&As[buf * ASZ], A + (size_t)bm * K + kb, 32, 128, K, 3, 7);
      tdm_load_2d(&Bs[buf * BSZ], BT + (size_t)bn * K + kb, 32, 128, K, 3, 7);
    }
#else
#pragma unroll
    for (int it = 0; it < 2; ++it) {
      int id = tid + it * 256;
      int r = id >> 2, cc = (id & 3) * 8;
      cp_lds16(&As[buf * ASZ + r * LDA + cc], A + (size_t)(bm + r) * K + kb + cc);
    }
#pragma unroll
    for (int it = 0; it < 2; ++it) {
      int id = tid + it * 256;
      int r = id >> 2, cc = (id & 3) * 8;
      cp_lds16(&Bs[buf * BSZ + r * LDB + cc], BT + (size_t)(bn + r) * K + kb + cc);
    }
#endif
  };

  v8f acc[2][4] = {};
  const int T = K / 32;
  stage(0, 0);
  for (int t = 0; t < T; ++t) {
    if (t + 1 < T) {
      stage((t + 1) & 1, (t + 1) * 32);
#if HAVE_TDM
      if (wid == 0) __builtin_amdgcn_s_wait_tensorcnt((short)2);  // leave next tile's 2 descs in flight
#else
      CP_WAIT(4);
#endif
    } else {
#if HAVE_TDM
      if (wid == 0) __builtin_amdgcn_s_wait_tensorcnt((short)0);
#else
      CP_WAIT(0);
#endif
    }
    __syncthreads();
    const half_t* Ab = &As[(t & 1) * ASZ];
    const half_t* Bb = &Bs[(t & 1) * BSZ];
    v16h af[2], bf[4];
#pragma unroll
    for (int i = 0; i < 2; ++i) af[i] = frag_a16(Ab, LDA, wm + 16 * i, 0, lane);
#pragma unroll
    for (int j = 0; j < 4; ++j) bf[j] = frag_bT(Bb, LDB, 0, wn + 16 * j, lane);
#pragma unroll
    for (int i = 0; i < 2; ++i)
#pragma unroll
      for (int j = 0; j < 4; ++j) acc[i][j] = wmma16(af[i], bf[j], acc[i][j]);
    __syncthreads();
  }

  const int col_in = lane & 15;
  const int rbase  = (lane >> 4) * 8;
#pragma unroll
  for (int i = 0; i < 2; ++i) {
#pragma unroll
    for (int j = 0; j < 4; ++j) {
      int row0 = bm + wm + 16 * i;
      int col0 = bn + wn + 16 * j;
#pragma unroll
      for (int r = 0; r < 8; ++r) {
        int row = row0 + rbase + r;
        int col = col0 + col_in;
        float v = acc[i][j][r];
        if (EPI == 0) {
          ((float*)Out)[(size_t)row * Nn + col] = v;
        } else if (EPI == 1) {
          v += bias[col];
          v = v > 0.f ? v : 0.f;
          ((half_t*)Out)[(size_t)row * Nn + col] = (half_t)v;
        } else {
          ((float*)Out)[(size_t)row * Nn + col] = v + bias[col];
        }
      }
    }
  }
}

// ---------------- conversions / packing ----------------
__global__ void cvt_f32_f16_k(const float* __restrict__ src, half_t* __restrict__ dst, size_t n) {
  size_t i = (size_t)blockIdx.x * blockDim.x + threadIdx.x;
  if (i < n) dst[i] = (half_t)src[i];
}
__global__ void transpose_f32_f16_k(const float* __restrict__ src, half_t* __restrict__ dst,
                                    int Rows, int Cols) {
  size_t i = (size_t)blockIdx.x * blockDim.x + threadIdx.x;
  if (i >= (size_t)Rows * Cols) return;
  int c = (int)(i / Rows), r = (int)(i % Rows);
  dst[i] = (half_t)src[(size_t)r * Cols + c];
}
__global__ void pack_wcatT_k(const float* __restrict__ Wqg, const float* __restrict__ Wkv,
                             half_t* __restrict__ wcatT) {
  size_t i = (size_t)blockIdx.x * blockDim.x + threadIdx.x;
  if (i >= (size_t)4 * C_ * C_) return;
  int nn = (int)(i / C_), k = (int)(i % C_);
  float v = (nn < 2 * C_) ? Wqg[(size_t)k * 2 * C_ + nn] : Wkv[(size_t)k * 2 * C_ + (nn - 2 * C_)];
  wcatT[i] = (half_t)v;
}

// ---------------- qgkv epilogue ----------------
__global__ void epi_qgkv_k(const float* __restrict__ qgkv, const float* __restrict__ pos,
                           const float* __restrict__ scale_p, const float* __restrict__ power_p,
                           half_t* __restrict__ qf, half_t* __restrict__ kcT,
                           half_t* __restrict__ vhT) {
  size_t idx = (size_t)blockIdx.x * blockDim.x + threadIdx.x;
  if (idx >= (size_t)MTOT * C_) return;
  int c = (int)(idx % C_);
  size_t bn = idx / C_;
  int n = (int)(bn % N_);
  int b = (int)(bn / N_);
  int h = c >> 6, d = c & 63;
  const float* row = qgkv + bn * (4 * C_);
  float q = row[c];
  float k = row[2 * C_ + c] + pos[(size_t)n * C_ + c];
  float v = row[3 * C_ + c];
  float sc = log1pf(expf(scale_p[c]));
  float pw = 1.0f + 4.0f / (1.0f + expf(-power_p[h * HD_ + d]));
  q /= sc; k /= sc;
  float qp = q > 0.f ? powf(q, pw) : 0.f;
  float qn = q < 0.f ? powf(-q, pw) : 0.f;
  float kp = k > 0.f ? powf(k, pw) : 0.f;
  float kn = k < 0.f ? powf(-k, pw) : 0.f;
  size_t bh = (size_t)b * NH_ + h;
  size_t o = (bh * N_ + n) * 128;
  qf[o + d]      = (half_t)qp;
  qf[o + 64 + d] = (half_t)qn;
  kcT[(bh * 128 + d) * N_ + n]      = (half_t)kp;
  kcT[(bh * 128 + 64 + d) * N_ + n] = (half_t)kn;
  vhT[(bh * 64 + d) * N_ + n]       = (half_t)v;
}

// ---------------- km = mean_n kc ----------------
__global__ __launch_bounds__(256) void km_reduce_k(const half_t* __restrict__ kcT, float* __restrict__ km) {
  int id = blockIdx.x;
  const half_t* base = kcT + (size_t)id * N_;
  float s = 0.f;
  for (int n = threadIdx.x; n < N_; n += 256) s += (float)base[n];
  __shared__ float red[256];
  red[threadIdx.x] = s;
  __syncthreads();
  for (int off = 128; off; off >>= 1) {
    if (threadIdx.x < off) red[threadIdx.x] += red[threadIdx.x + off];
    __syncthreads();
  }
  if (threadIdx.x == 0) km[id] = red[0] / (float)N_;
}

// ---------------- z factors ----------------
__global__ void z_k(const half_t* __restrict__ qf, const float* __restrict__ km,
                    float* __restrict__ zbuf) {
  int idx = blockIdx.x * blockDim.x + threadIdx.x;
  if (idx >= B_ * NH_ * N_) return;
  int bh = idx / N_;
  const half_t* qrow = qf + (size_t)idx * 128;
  const float* kmb = km + (size_t)bh * 128;
  float d0 = 0.f, d1 = 0.f;
#pragma unroll 4
  for (int d = 0; d < 64; ++d) {
    float a = (float)qrow[d], bq = (float)qrow[64 + d];
    float k0 = kmb[d], k1 = kmb[64 + d];
    d0 += a * k0 + bq * k1;
    d1 += bq * k0 + a * k1;
  }
  zbuf[(size_t)idx * 2]     = 1.f / (d0 + 1e-6f);
  zbuf[(size_t)idx * 2 + 1] = 1.f / (d1 + 1e-6f);
}

// ---------------- kvagg^T per (b,h), async-staged (ASYNCcnt path), double-buffered ----------------
__global__ __launch_bounds__(256) void kvagg_k(const half_t* __restrict__ kcT,
                                               const half_t* __restrict__ vhT,
                                               half_t* __restrict__ kvT) {
  constexpr int LDA = 48, LDB = 48;
  constexpr int ASZ = 128 * LDA, BSZ = 64 * LDB;
  __shared__ half_t As[2 * ASZ];
  __shared__ half_t Bs[2 * BSZ];
  const int bh = blockIdx.x;
  const int tid = threadIdx.x, lane = tid & 31, wid = tid >> 5;
  const int wrow = wid * 16;
  const half_t* Abase = kcT + (size_t)bh * 128 * N_;
  const half_t* Bbase = vhT + (size_t)bh * 64 * N_;

  auto stage = [&](int buf, int nb) {
#pragma unroll
    for (int it = 0; it < 2; ++it) {
      int id = tid + it * 256;
      int r = id >> 2, cc = (id & 3) * 8;
      cp_lds16(&As[buf * ASZ + r * LDA + cc], Abase + (size_t)r * N_ + nb + cc);
    }
    {
      int r = tid >> 2, cc = (tid & 3) * 8;
      cp_lds16(&Bs[buf * BSZ + r * LDB + cc], Bbase + (size_t)r * N_ + nb + cc);
    }
  };

  v8f acc[4] = {};
  const int T = N_ / 32;
  stage(0, 0);
  for (int t = 0; t < T; ++t) {
    if (t + 1 < T) {
      stage((t + 1) & 1, (t + 1) * 32);
      CP_WAIT(3);
    } else {
      CP_WAIT(0);
    }
    __syncthreads();
    const half_t* Ab = &As[(t & 1) * ASZ];
    const half_t* Bb = &Bs[(t & 1) * BSZ];
    v16h a = frag_a16(Ab, LDA, wrow, 0, lane);
#pragma unroll
    for (int j = 0; j < 4; ++j) {
      v16h bf = frag_bT(Bb, LDB, 0, 16 * j, lane);
      acc[j] = wmma16(a, bf, acc[j]);
    }
    __syncthreads();
  }

  const float s2 = 1.0f / (float)N_;
  half_t* outb = kvT + (size_t)bh * 64 * 128;
  const int col_in = lane & 15, rbase = (lane >> 4) * 8;
#pragma unroll
  for (int j = 0; j < 4; ++j) {
#pragma unroll
    for (int r = 0; r < 8; ++r) {
      int dd  = wrow + rbase + r;
      int col = 16 * j + col_in;
      int od  = (col < 32) ? dd : ((dd + 64) & 127);
      outb[(size_t)col * 128 + od] = (half_t)(acc[j][r] * s2);
    }
  }
}

// ---------------- attention GEMM, TDM-staged, z epilogue ----------------
__global__ __launch_bounds__(256) void attn_gemm_k(const half_t* __restrict__ qf,
                                                   const half_t* __restrict__ kvT,
                                                   const float* __restrict__ zbuf,
                                                   float* __restrict__ attn) {
  constexpr int LDA = 144, LDB = 144;     // 128 data + 16 pad halfs (TDM: pad 8 DW per 64 DW)
  __shared__ half_t Als[64 * LDA];
  __shared__ half_t Bls[64 * LDB];
  const int bh = blockIdx.y;
  const int b = bh >> 3, h = bh & 7;
  const int nb = blockIdx.x * 64;
  const int tid = threadIdx.x, lane = tid & 31, wid = tid >> 5;
  const int wm = (wid >> 1) * 16, wn = (wid & 1) * 32;
  const half_t* Bbase = kvT + (size_t)bh * 64 * 128;
  const half_t* Abase = qf + ((size_t)bh * N_ + nb) * 128;

#if HAVE_TDM
  if (wid == 0) {
    tdm_load_2d(Als, Abase, 64 * 128, 1, 64 * 128, 5, 7);   // 16KB contiguous, padded pitch 144
    tdm_load_2d(Bls, Bbase, 64 * 128, 1, 64 * 128, 5, 7);
    __builtin_amdgcn_s_wait_tensorcnt((short)0);
  }
#else
#pragma unroll
  for (int j = 0; j < 4; ++j) {
    int chunk = tid * 4 + j;
    int r = chunk >> 4, c8 = (chunk & 15) * 8;
    cp_lds16(&Als[r * LDA + c8], Abase + (size_t)r * 128 + c8);
  }
#pragma unroll
  for (int j = 0; j < 4; ++j) {
    int chunk = tid * 4 + j;
    int r = chunk >> 4, c8 = (chunk & 15) * 8;
    cp_lds16(&Bls[r * LDB + c8], Bbase + (size_t)r * 128 + c8);
  }
  CP_WAIT(0);
#endif
  __syncthreads();

  v8f acc[2] = {};
#pragma unroll
  for (int ks = 0; ks < 128; ks += 32) {
    v16h a = frag_a16(Als, LDA, wm, ks, lane);
#pragma unroll
    for (int j = 0; j < 2; ++j) {
      v16h bf = frag_bT(Bls, LDB, ks, wn + 16 * j, lane);
      acc[j] = wmma16(a, bf, acc[j]);
    }
  }

  const int col_in = lane & 15, rbase = (lane >> 4) * 8;
#pragma unroll
  for (int j = 0; j < 2; ++j) {
#pragma unroll
    for (int r = 0; r < 8; ++r) {
      int nrow = nb + wm + rbase + r;
      int col  = wn + 16 * j + col_in;
      float z  = zbuf[((size_t)bh * N_ + nrow) * 2 + (col >> 5)];
      attn[((size_t)b * N_ + nrow) * C_ + h * 64 + col] = acc[j][r] * z;
    }
  }
}

// ---------------- 5x5 depthwise conv over v ----------------
__global__ void dwconv_k(const float* __restrict__ qgkv, const float* __restrict__ w,
                         const float* __restrict__ bias, float* __restrict__ vout) {
  size_t idx = (size_t)blockIdx.x * blockDim.x + threadIdx.x;
  if (idx >= (size_t)MTOT * C_) return;
  int c = (int)(idx % C_);
  size_t bn = idx / C_;
  int n = (int)(bn % N_);
  int b = (int)(bn / N_);
  int y = n / W_, x = n % W_;
  int d = c & 63;
  float acc = bias[d];
#pragma unroll
  for (int ky = 0; ky < KS_; ++ky) {
    int iy = y + ky - 2;
    if (iy < 0 || iy >= H_) continue;
#pragma unroll
    for (int kx = 0; kx < KS_; ++kx) {
      int ix = x + kx - 2;
      if (ix < 0 || ix >= W_) continue;
      acc += w[d * 25 + ky * 5 + kx] *
             qgkv[((size_t)b * N_ + iy * W_ + ix) * (4 * C_) + 3 * C_ + c];
    }
  }
  vout[idx] = acc;
}

// ---------------- x2 = (attn + vout) * g ----------------
__global__ void x2_k(float* __restrict__ attn, const float* __restrict__ vout,
                     const float* __restrict__ qgkv, half_t* __restrict__ x2h) {
  size_t idx = (size_t)blockIdx.x * blockDim.x + threadIdx.x;
  if (idx >= (size_t)MTOT * C_) return;
  int c = (int)(idx % C_);
  size_t bn = idx / C_;
  float g = qgkv[bn * (4 * C_) + C_ + c];
  float v = (attn[idx] + vout[idx]) * g;
  attn[idx] = v;
  x2h[idx] = (half_t)v;
}

// ---------------- residual + LayerNorm ----------------
__global__ __launch_bounds__(256) void ln_k(const float* __restrict__ x2, const float* __restrict__ ff,
                                            const float* __restrict__ lng, const float* __restrict__ lnb,
                                            float* __restrict__ out) {
  int row = blockIdx.x;
  const float* a = x2 + (size_t)row * C_;
  const float* f = ff + (size_t)row * C_;
  __shared__ float yv[C_];
  __shared__ float r1[256];
  __shared__ float r2[256];
  int tid = threadIdx.x;
  float s = 0.f, sq = 0.f;
  for (int c = tid; c < C_; c += 256) {
    float v = a[c] + f[c];
    yv[c] = v;
    s += v; sq += v * v;
  }
  r1[tid] = s; r2[tid] = sq;
  __syncthreads();
  for (int off = 128; off; off >>= 1) {
    if (tid < off) { r1[tid] += r1[tid + off]; r2[tid] += r2[tid + off]; }
    __syncthreads();
  }
  float mean = r1[0] / (float)C_;
  float var  = r2[0] / (float)C_ - mean * mean;
  float rstd = rsqrtf(var + 1e-5f);
  for (int c = tid; c < C_; c += 256)
    out[(size_t)row * C_ + c] = (yv[c] - mean) * rstd * lng[c] + lnb[c];
}

// ---------------- host launch ----------------
extern "C" void kernel_launch(void* const* d_in, const int* in_sizes, int n_in,
                              void* d_out, int out_size, void* d_ws, size_t ws_size,
                              hipStream_t stream) {
  const float* x       = (const float*)d_in[0];
  const float* Wqg     = (const float*)d_in[1];
  const float* Wkv     = (const float*)d_in[2];
  const float* pos     = (const float*)d_in[3];
  const float* scale_p = (const float*)d_in[4];
  const float* power_p = (const float*)d_in[5];
  const float* dwc_w   = (const float*)d_in[6];
  const float* dwc_b   = (const float*)d_in[7];
  const float* ln_g    = (const float*)d_in[8];
  const float* ln_b    = (const float*)d_in[9];
  const float* W1      = (const float*)d_in[10];
  const float* b1      = (const float*)d_in[11];
  const float* W2      = (const float*)d_in[12];
  const float* b2      = (const float*)d_in[13];
  float* out = (float*)d_out;

  char* ws = (char*)d_ws;
  half_t* xh    = (half_t*)(ws + OFF_XH);
  float*  qgkv  = (float*)(ws + OFF_QGKV);
  half_t* hh    = (half_t*)(ws + OFF_QGKV);
  half_t* wcatT = (half_t*)(ws + OFF_WCAT);
  half_t* w1T   = (half_t*)(ws + OFF_W1T);
  half_t* w2T   = (half_t*)(ws + OFF_W2T);
  half_t* qf    = (half_t*)(ws + OFF_QF);
  half_t* kcT   = (half_t*)(ws + OFF_KCT);
  half_t* vhT   = (half_t*)(ws + OFF_VHT);
  float*  km    = (float*)(ws + OFF_KM);
  half_t* kvT   = (half_t*)(ws + OFF_KVT);
  float*  zbuf  = (float*)(ws + OFF_Z);
  float*  attn  = (float*)(ws + OFF_ATTN);
  float*  vout  = (float*)(ws + OFF_VOUT);
  float*  ff    = (float*)(ws + OFF_VOUT);

  const size_t nElem = (size_t)MTOT * C_;
  const int blk = 256;
  const unsigned gElem = (unsigned)((nElem + blk - 1) / blk);

  cvt_f32_f16_k<<<gElem, blk, 0, stream>>>(x, xh, nElem);
  transpose_f32_f16_k<<<(unsigned)(((size_t)C_ * FF_ + blk - 1) / blk), blk, 0, stream>>>(W1, w1T, C_, FF_);
  transpose_f32_f16_k<<<(unsigned)(((size_t)FF_ * C_ + blk - 1) / blk), blk, 0, stream>>>(W2, w2T, FF_, C_);
  pack_wcatT_k<<<(unsigned)(((size_t)C_ * 4 * C_ + blk - 1) / blk), blk, 0, stream>>>(Wqg, Wkv, wcatT);

  gemm_f16_k<0><<<dim3(MTOT / 128, (4 * C_) / 128), 256, 0, stream>>>(xh, wcatT, qgkv, nullptr, MTOT, 4 * C_, C_);

  epi_qgkv_k<<<gElem, blk, 0, stream>>>(qgkv, pos, scale_p, power_p, qf, kcT, vhT);

  km_reduce_k<<<B_ * NH_ * 128, 256, 0, stream>>>(kcT, km);
  z_k<<<(B_ * NH_ * N_ + blk - 1) / blk, blk, 0, stream>>>(qf, km, zbuf);

  kvagg_k<<<B_ * NH_, 256, 0, stream>>>(kcT, vhT, kvT);

  attn_gemm_k<<<dim3(N_ / 64, B_ * NH_), 256, 0, stream>>>(qf, kvT, zbuf, attn);

  dwconv_k<<<gElem, blk, 0, stream>>>(qgkv, dwc_w, dwc_b, vout);
  x2_k<<<gElem, blk, 0, stream>>>(attn, vout, qgkv, xh);

  gemm_f16_k<1><<<dim3(MTOT / 128, FF_ / 128), 256, 0, stream>>>(xh, w1T, hh, b1, MTOT, FF_, C_);
  gemm_f16_k<2><<<dim3(MTOT / 128, C_ / 128), 256, 0, stream>>>(hh, w2T, ff, b2, MTOT, C_, FF_);

  ln_k<<<MTOT, 256, 0, stream>>>(attn, ff, ln_g, ln_b, out);
}
